// NewUnevenRTrainer_85461259256171
// MI455X (gfx1250) — matrix operand order repaired and checked by reading, not compiled, not run on hardware
//
#include <hip/hip_runtime.h>

typedef __attribute__((ext_vector_type(16))) _Float16 v16h;
typedef __attribute__((ext_vector_type(8)))  _Float16 v8h;
typedef __attribute__((ext_vector_type(4)))  _Float16 v4h;
typedef __attribute__((ext_vector_type(8)))  float    v8f;
typedef unsigned int v4u __attribute__((ext_vector_type(4)));
typedef int          v8i __attribute__((ext_vector_type(8)));
typedef int          v4i __attribute__((ext_vector_type(4)));

#define HD 1024

#if __has_builtin(__builtin_amdgcn_tensor_load_to_lds) && \
    __has_builtin(__builtin_amdgcn_s_wait_tensorcnt)
#define HAVE_TDM 1
#else
#define HAVE_TDM 0
#endif

union V16U { v16h v; v8h h[2]; };

__device__ __forceinline__ v16h join8(v8h lo, v8h hi) {
  V16U u; u.h[0] = lo; u.h[1] = hi; return u.v;
}

// A fragment (16x32 f16 striping) built from an f32 LDS row.
__device__ __forceinline__ v16h afrag_f32(const float* arow, int koffA) {
  float4 f0 = *(const float4*)(arow + koffA);
  float4 f1 = *(const float4*)(arow + koffA + 4);
  float4 f2 = *(const float4*)(arow + koffA + 16);
  float4 f3 = *(const float4*)(arow + koffA + 20);
  v16h a;
  a[0]  = (_Float16)f0.x; a[1]  = (_Float16)f0.y; a[2]  = (_Float16)f0.z; a[3]  = (_Float16)f0.w;
  a[4]  = (_Float16)f1.x; a[5]  = (_Float16)f1.y; a[6]  = (_Float16)f1.z; a[7]  = (_Float16)f1.w;
  a[8]  = (_Float16)f2.x; a[9]  = (_Float16)f2.y; a[10] = (_Float16)f2.z; a[11] = (_Float16)f2.w;
  a[12] = (_Float16)f3.x; a[13] = (_Float16)f3.y; a[14] = (_Float16)f3.z; a[15] = (_Float16)f3.w;
  return a;
}

// Column -> unit mapping: groups (256,2),(128,3),(64,2)
__device__ __forceinline__ int unit_of_col(int c) {
  if (c < 512) return c >> 8;
  if (c < 896) return 2 + ((c - 512) >> 7);
  return 5 + ((c - 896) >> 6);
}

__device__ __forceinline__ float block_sum(float x, volatile float* sm) {
  #pragma unroll
  for (int m = 16; m; m >>= 1) x += __shfl_xor(x, m, 32);
  int w = threadIdx.x >> 5;
  __syncthreads();
  if ((threadIdx.x & 31) == 0) sm[w] = x;
  __syncthreads();
  if (threadIdx.x == 0) {
    float r = 0.f;
    int nw = blockDim.x >> 5;
    for (int i = 0; i < nw; ++i) r += sm[i];
    sm[0] = r;
  }
  __syncthreads();
  return sm[0];
}

#if HAVE_TDM
// ---------------------------------------------------------------------------
// Tensor Data Mover: 2-D tile load global -> LDS with LDS padding.
// D# per CDNA5 ISA 8.3/8.4. data_size code: 0=1B,1=2B,2=4B,3=8B.
// pad_interval code: 2^(c+1) dwords between pads; pad_amount code: c+1 dwords.
// ---------------------------------------------------------------------------
__device__ __forceinline__ void tdm_load_2d(
    const void* gtile, unsigned lds_addr, unsigned ds_code,
    unsigned td0, unsigned td1, unsigned tile0, unsigned tile1,
    unsigned stride0, unsigned pad_interval, unsigned pad_amount)
{
  unsigned long long ga = (unsigned long long)gtile;
  v4u g0;
  g0[0] = 1u;                                          // count=1, user mode
  g0[1] = lds_addr;                                    // LDS byte address
  g0[2] = (unsigned)(ga & 0xFFFFFFFFu);                // global_addr[31:0]
  g0[3] = (unsigned)((ga >> 32) & 0x1FFFFFFu) | (2u << 30);  // [56:32] | type=2
  v8i g1;
  g1[0] = (int)((ds_code << 16) | (1u << 20) |
                (pad_interval << 22) | (pad_amount << 25)); // pad_enable=1
  g1[1] = (int)((td0 & 0xFFFFu) << 16);                // tensor_dim0 lo16
  g1[2] = (int)(((td0 >> 16) & 0xFFFFu) | ((td1 & 0xFFFFu) << 16));
  g1[3] = (int)(((td1 >> 16) & 0xFFFFu) | (tile0 << 16));
  g1[4] = (int)(tile1 & 0xFFFFu);                      // tile_dim1, tile_dim2=0
  g1[5] = (int)stride0;                                // dim0 stride lo32
  g1[6] = 0;
  g1[7] = 0;
  v4i z4; z4[0] = 0; z4[1] = 0; z4[2] = 0; z4[3] = 0;
#if __clang_major__ >= 23
  v8i z8; for (int i = 0; i < 8; ++i) z8[i] = 0;
  __builtin_amdgcn_tensor_load_to_lds(g0, g1, z4, z4, z8, 0);
#else
  __builtin_amdgcn_tensor_load_to_lds(g0, g1, z4, z4, 0);
#endif
}
#endif  // HAVE_TDM

// ---------------------------------------------------------------------------
// rot = f16( X[f32, M x 1024] @ R ) via R16T (pre-transposed f16 R), plus
// per-subspace sum-of-squares into norm2[7][M] via atomics.
// 256 threads (8 waves), tile 64(M) x 256(N); wave = 16x128.
// TDM double-buffered staging; LDS: A f32 [64][36], B f16 [256][40].
// ---------------------------------------------------------------------------
__global__ __launch_bounds__(256) void rot_wmma_kernel(
    const float* __restrict__ X, const _Float16* __restrict__ R16T,
    _Float16* __restrict__ out16, float* __restrict__ norm2, int M)
{
  __shared__ float    At[2][64 * 36];
  __shared__ _Float16 Bt[2][256 * 40];

  const int t = threadIdx.x;
  const int lane = t & 31, w = t >> 5;
  const int wm = (w & 3) * 16;
  const int wn = (w >> 2) * 128;
  const int mbase = blockIdx.x * 64;
  const int nbase = blockIdx.y * 256;
  const int row   = lane & 15;
  const int hi    = (lane >> 4) & 1;
  const int koffA = hi << 3;
  const int koffB = hi << 3;
  const int moff  = hi << 3;

  v8f acc[8];
  #pragma unroll
  for (int s = 0; s < 8; ++s)
    #pragma unroll
    for (int i = 0; i < 8; ++i) acc[s][i] = 0.f;

  const int NIT = HD / 32;

#if HAVE_TDM
  if (w == 0) {
    tdm_load_2d(X + (size_t)mbase * HD, (unsigned)(unsigned long long)&At[0][0],
                2, HD, (unsigned)M, 32, 64, HD, 4, 3);
    tdm_load_2d(R16T + (size_t)nbase * HD, (unsigned)(unsigned long long)&Bt[0][0],
                1, HD, HD, 32, 256, HD, 3, 3);
  }
#endif

  for (int it = 0; it < NIT; ++it) {
    const int p = it & 1;
#if HAVE_TDM
    if (w == 0) {
      if (it + 1 < NIT) {
        const int k0n = (it + 1) * 32;
        tdm_load_2d(X + (size_t)mbase * HD + k0n,
                    (unsigned)(unsigned long long)&At[p ^ 1][0],
                    2, HD, (unsigned)M, 32, 64, HD, 4, 3);
        tdm_load_2d(R16T + (size_t)nbase * HD + k0n,
                    (unsigned)(unsigned long long)&Bt[p ^ 1][0],
                    1, HD, HD, 32, 256, HD, 3, 3);
        __builtin_amdgcn_s_wait_tensorcnt(2);
      } else {
        __builtin_amdgcn_s_wait_tensorcnt(0);
      }
    }
    __syncthreads();
#else
    {
      const int k0 = it * 32;
      int r = t >> 2, c = (t & 3) * 8;
      const float4* asrc = (const float4*)(X + (size_t)(mbase + r) * HD + k0 + c);
      *(float4*)(&At[p][r * 36 + c])     = asrc[0];
      *(float4*)(&At[p][r * 36 + c + 4]) = asrc[1];
      const _Float16* bsrc = R16T + (size_t)(nbase + t) * HD + k0;
      #pragma unroll
      for (int i = 0; i < 4; ++i)
        *(v8h*)(&Bt[p][t * 40 + i * 8]) = *(const v8h*)(bsrc + i * 8);
    }
    __syncthreads();
#endif

    v16h a = afrag_f32(&At[p][(wm + row) * 36], koffA);
    #pragma unroll
    for (int s = 0; s < 8; ++s) {
      const _Float16* bp = &Bt[p][(wn + s * 16 + row) * 40 + koffB];
      v16h b = join8(*(const v8h*)bp, *(const v8h*)(bp + 16));
      acc[s] = __builtin_amdgcn_wmma_f32_16x16x32_f16(
          false, a, false, b, (short)0, acc[s], false, false);
    }
    __syncthreads();
  }

  #pragma unroll
  for (int s = 0; s < 8; ++s) {
    int n = nbase + wn + s * 16 + row;
    int u = unit_of_col(n);
    float sq[8];
    #pragma unroll
    for (int v = 0; v < 8; ++v) {
      int m = mbase + wm + v + moff;
      float x = acc[s][v];
      out16[(size_t)m * HD + n] = (_Float16)x;
      sq[v] = x * x;
    }
    #pragma unroll
    for (int msk = 1; msk <= 8; msk <<= 1)
      #pragma unroll
      for (int v = 0; v < 8; ++v) sq[v] += __shfl_xor(sq[v], msk, 32);
    if (row == 0) {
      #pragma unroll
      for (int v = 0; v < 8; ++v)
        atomicAdd(&norm2[(size_t)u * M + (mbase + wm + v + moff)], sq[v]);
    }
  }
}

// ---------------------------------------------------------------------------
// sim = Qn[rows x D] @ rot_subspace^T scaled by 1/clip(||k||,1e-3).
// Grid: (row tiles, key segments). Each workgroup owns (tile, segment):
// 4 waves split the segment, running (max, argmax), merged into the
// per-segment partial slot (race-free per launch; chunks serialize).
// ---------------------------------------------------------------------------
template<int D, int KS>
__global__ __launch_bounds__(128) void sim_wmma_kernel(
    const _Float16* __restrict__ qn, const _Float16* __restrict__ rot16,
    const float* __restrict__ invn, float* __restrict__ pval,
    int* __restrict__ pidx, int chunk, int segsz, int chunk_base,
    int unit_off, int gcol)
{
  static_assert(D == KS * 32, "D/KS mismatch");
  __shared__ float rvs[4][16];
  __shared__ int   ris[4][16];

  const int t = threadIdx.x, lane = t & 31, w = t >> 5;
  const int rbase = blockIdx.x * 16;
  const int row   = lane & 15;
  const int hi    = (lane >> 4) & 1;
  const int koffA = hi << 3;
  const int koffB = hi << 3;
  const int moff  = hi << 3;
  const int u     = unit_off + (rbase >> 8);

  v16h afr[KS];
  {
    const _Float16* qrow = qn + (size_t)(rbase + row) * D;
    #pragma unroll
    for (int ks = 0; ks < KS; ++ks)
      afr[ks] = join8(*(const v8h*)(qrow + ks * 32 + koffA),
                      *(const v8h*)(qrow + ks * 32 + koffA + 16));
  }

  float mv[8]; int mi[8];
  #pragma unroll
  for (int v = 0; v < 8; ++v) { mv[v] = -3.0e38f; mi[v] = -1; }

  const int kseg = segsz >> 2;                 // keys per wave
  const int l0 = blockIdx.y * segsz + w * kseg;
  for (int l = l0; l < l0 + kseg; l += 16) {
    int kc = l + row;
    const _Float16* krow = rot16 + (size_t)kc * HD + gcol;
    v8f acc;
    #pragma unroll
    for (int i = 0; i < 8; ++i) acc[i] = 0.f;
    #pragma unroll
    for (int ks = 0; ks < KS; ++ks) {
      v16h b = join8(*(const v8h*)(krow + ks * 32 + koffB),
                     *(const v8h*)(krow + ks * 32 + koffB + 16));
      acc = __builtin_amdgcn_wmma_f32_16x16x32_f16(
          false, afr[ks], false, b, (short)0, acc, false, false);
    }
    float inv = invn[(size_t)u * chunk + kc];
    int gi = chunk_base + kc;
    #pragma unroll
    for (int v = 0; v < 8; ++v) {
      float val = acc[v] * inv;
      if (val > mv[v]) { mv[v] = val; mi[v] = gi; }  // strict > keeps first
    }
  }

  #pragma unroll
  for (int msk = 1; msk <= 8; msk <<= 1) {
    #pragma unroll
    for (int v = 0; v < 8; ++v) {
      float ov = __shfl_xor(mv[v], msk, 32);
      int   oi = __shfl_xor(mi[v], msk, 32);
      if (ov > mv[v] ||
          (ov == mv[v] && oi != -1 && (mi[v] == -1 || oi < mi[v]))) {
        mv[v] = ov; mi[v] = oi;
      }
    }
  }
  if (row == 0) {
    #pragma unroll
    for (int v = 0; v < 8; ++v) { rvs[w][v + moff] = mv[v]; ris[w][v + moff] = mi[v]; }
  }
  __syncthreads();
  if (t < 16) {
    float bv = rvs[0][t]; int bi = ris[0][t];
    #pragma unroll
    for (int ww = 1; ww < 4; ++ww) {
      float ov = rvs[ww][t]; int oi = ris[ww][t];
      if (ov > bv || (ov == bv && oi != -1 && (bi == -1 || oi < bi))) {
        bv = ov; bi = oi;
      }
    }
    size_t slot = (size_t)blockIdx.y * 1792 + u * 256 + ((rbase + t) & 255);
    float cur = pval[slot]; int ci = pidx[slot];
    if (bv > cur || (bv == cur && bi != -1 && (ci == -1 || bi < ci))) {
      pval[slot] = bv; pidx[slot] = bi;
    }
  }
}

// Fold per-segment partials into the global best (value desc, index asc),
// floored at -1 / idx=-1 to match the reference scan init.
__global__ void merge_kernel(const float* __restrict__ pval,
                             const int* __restrict__ pidx, int nseg,
                             float* __restrict__ bval, int* __restrict__ bidx) {
  int off = blockIdx.x * blockDim.x + threadIdx.x;
  if (off >= 1792) return;
  float bv = -1.f; int bi = -1;
  for (int s = 0; s < nseg; ++s) {
    float v = pval[s * 1792 + off]; int i = pidx[s * 1792 + off];
    if (v > bv || (v == bv && i != -1 && (bi == -1 || i < bi))) { bv = v; bi = i; }
  }
  bval[off] = bv; bidx[off] = bi;
}

// ---------------------------------------------------------------------------
// q_normed rows (group-major, 1792 rows) from rotated query, emitted as f16.
// ---------------------------------------------------------------------------
__global__ __launch_bounds__(128) void qnorm_kernel(
    const _Float16* __restrict__ qr16, _Float16* __restrict__ qn16)
{
  __shared__ float sm[8];
  int r = blockIdx.x;
  int d, coloff, qbase, rl;
  if (r < 512)       { d = 256; coloff = 0;   qbase = 0;      rl = r; }
  else if (r < 1280) { d = 128; coloff = 512; qbase = 131072; rl = r - 512; }
  else               { d = 64;  coloff = 896; qbase = 229376; rl = r - 1280; }
  int ul = rl >> 8, b = rl & 255;
  int c0 = coloff + ul * d;
  float ss = 0.f;
  for (int j = threadIdx.x; j < d; j += 128) {
    float q = (float)qr16[(size_t)b * HD + c0 + j];
    ss += q * q;
  }
  float tot = block_sum(ss, sm);
  float scale = 1.f / fmaxf(sqrtf(tot), 1e-8f);
  for (int j = threadIdx.x; j < d; j += 128) {
    float q = (float)qr16[(size_t)b * HD + c0 + j];
    qn16[(size_t)qbase + (size_t)rl * d + j] = (_Float16)(q * scale);
  }
}

// ---------------------------------------------------------------------------
// Finalize: recompute best key's rotated subspace in f32; cosine vs rotated
// query subspace; accumulate -d*cos/(256*1024) into scalar output.
// ---------------------------------------------------------------------------
__global__ __launch_bounds__(256) void finalize_kernel(
    const float* __restrict__ keys, const float* __restrict__ R,
    const _Float16* __restrict__ qr16, const int* __restrict__ best_idx,
    float* __restrict__ out)
{
  __shared__ float sm[8];
  int r = blockIdx.x;
  int d, coloff, rl, uoff;
  if (r < 512)       { d = 256; coloff = 0;   rl = r;        uoff = 0; }
  else if (r < 1280) { d = 128; coloff = 512; rl = r - 512;  uoff = 2; }
  else               { d = 64;  coloff = 896; rl = r - 1280; uoff = 5; }
  int ul = rl >> 8, b = rl & 255;
  int u = uoff + ul;
  int c0 = coloff + ul * d;
  int idx = best_idx[u * 256 + b];

  int j = threadIdx.x;
  float diag = 0.f, qj = 0.f;
  if (j < d) {
    int col = c0 + j;
    if (idx >= 0) {
      const float* kv = keys + (size_t)idx * HD;
      float s = 0.f;
      for (int k = 0; k < HD; ++k) s += kv[k] * R[(size_t)k * HD + col];
      diag = s;
    }
    qj = (float)qr16[(size_t)b * HD + col];
  }
  float d2 = block_sum(diag * diag, sm);
  float qd = block_sum(qj * diag, sm);
  float q2 = block_sum(qj * qj, sm);
  if (threadIdx.x == 0) {
    float cosv = qd / (fmaxf(sqrtf(q2), 1e-8f) * fmaxf(sqrtf(d2), 1e-8f));
    atomicAdd(out, -cosv * (float)d / (256.f * 1024.f));
  }
}

// ------------------------------- utilities ---------------------------------
__global__ __launch_bounds__(256) void transpose_cvt_kernel(
    const float* __restrict__ R, _Float16* __restrict__ R16T)
{
  __shared__ _Float16 tile[32 * 36];
  const int bi = blockIdx.x * 32;   // k base
  const int bj = blockIdx.y * 32;   // n base
  {
    int kk = threadIdx.x >> 3, nn = (threadIdx.x & 7) * 4;
    float4 f = *(const float4*)(R + (size_t)(bi + kk) * HD + bj + nn);
    v4h h; h[0] = (_Float16)f.x; h[1] = (_Float16)f.y;
           h[2] = (_Float16)f.z; h[3] = (_Float16)f.w;
    *(v4h*)(&tile[kk * 36 + nn]) = h;
  }
  __syncthreads();
  {
    int nn = threadIdx.x >> 3, kk = (threadIdx.x & 7) * 4;
    v4h h;
    #pragma unroll
    for (int q = 0; q < 4; ++q) h[q] = tile[(kk + q) * 36 + nn];
    *(v4h*)(R16T + (size_t)(bj + nn) * HD + bi + kk) = h;
  }
}
__global__ void init_kernel(float* bv, int* bi, float* pv, int* pi, float* out) {
  int i = blockIdx.x * blockDim.x + threadIdx.x;
  if (i < 1792) { bv[i] = -1.f; bi[i] = -1; }
  if (i < 8 * 1792) { pv[i] = -3.0e38f; pi[i] = -1; }
  if (i == 0) out[0] = 0.f;
}
__global__ void zero_kernel(float* p, int n) {
  int i = blockIdx.x * blockDim.x + threadIdx.x;
  if (i < n) p[i] = 0.f;
}
__global__ void invnorm_kernel(float* p, int n) {
  int i = blockIdx.x * blockDim.x + threadIdx.x;
  if (i < n) p[i] = 1.f / fmaxf(sqrtf(fmaxf(p[i], 0.f)), 1e-3f);
}

// ---------------------------------------------------------------------------
extern "C" void kernel_launch(void* const* d_in, const int* in_sizes, int n_in,
                              void* d_out, int out_size, void* d_ws,
                              size_t ws_size, hipStream_t stream) {
  const float* query = (const float*)d_in[0];
  const float* keys  = (const float*)d_in[1];
  const float* R     = (const float*)d_in[2];
  float* out = (float*)d_out;
  const int NK = in_sizes[1] / HD;   // total keys (131072)

  char* ws = (char*)d_ws;
  _Float16* R16T = (_Float16*)(ws + 0);         // 2 MB
  _Float16* qr16 = (_Float16*)(ws + 2097152);   // 512 KB
  _Float16* qn16 = (_Float16*)(ws + 2621440);   // 512 KB
  float*    qn2  = (float*)   (ws + 3145728);   // 7*256*4 scratch (query rot)
  float*    bval = (float*)   (ws + 3152896);   // 7*256*4
  int*      bidx = (int*)     (ws + 3160064);   // 7*256*4
  float*    pval = (float*)   (ws + 3167232);   // 8*1792*4 partial best vals
  int*      pidx = (int*)     (ws + 3224576);   // 8*1792*4 partial best idxs
  const size_t dyn = 3281920;

  int CH = 1024;
  const int cands[5] = {32768, 16384, 8192, 4096, 2048};
  for (int i = 0; i < 5; ++i) {
    size_t need = dyn + (size_t)cands[i] * 28 + 256 + (size_t)cands[i] * 2048;
    if (need <= ws_size && (NK % cands[i]) == 0) { CH = cands[i]; break; }
  }
  while (CH > 64 && (NK % CH)) CH >>= 1;
  const int nseg = (CH >= 4096) ? (CH / 4096) : 1;   // <= 8
  const int segsz = CH / nseg;

  float* norm2 = (float*)(ws + dyn);
  size_t rot_off = (dyn + (size_t)CH * 28 + 255) & ~(size_t)255;
  _Float16* rot16 = (_Float16*)(ws + rot_off);

  transpose_cvt_kernel<<<dim3(32, 32), 256, 0, stream>>>(R, R16T);
  init_kernel<<<56, 256, 0, stream>>>(bval, bidx, pval, pidx, out);

  // Rotated query (reuse the WMMA GEMM kernel; qn2 norms unused).
  rot_wmma_kernel<<<dim3(4, 4), 256, 0, stream>>>(query, R16T, qr16, qn2, 256);
  qnorm_kernel<<<1792, 128, 0, stream>>>(qr16, qn16);

  for (int c = 0; c < NK; c += CH) {
    zero_kernel<<<(7 * CH + 255) / 256, 256, 0, stream>>>(norm2, 7 * CH);
    rot_wmma_kernel<<<dim3(CH / 64, 4), 256, 0, stream>>>(
        keys + (size_t)c * HD, R16T, rot16, norm2, CH);
    invnorm_kernel<<<(7 * CH + 255) / 256, 256, 0, stream>>>(norm2, 7 * CH);
    sim_wmma_kernel<256, 8><<<dim3(32, nseg), 128, 0, stream>>>(
        qn16 + 0,      rot16, norm2, pval, pidx, CH, segsz, c, 0, 0);
    sim_wmma_kernel<128, 4><<<dim3(48, nseg), 128, 0, stream>>>(
        qn16 + 131072, rot16, norm2, pval, pidx, CH, segsz, c, 2, 512);
    sim_wmma_kernel< 64, 2><<<dim3(32, nseg), 128, 0, stream>>>(
        qn16 + 229376, rot16, norm2, pval, pidx, CH, segsz, c, 5, 896);
  }

  merge_kernel<<<7, 256, 0, stream>>>(pval, pidx, nseg, bval, bidx);
  finalize_kernel<<<1792, 256, 0, stream>>>(keys, R, qr16, bidx, out);
}
